// MainNetwork_39616778338374
// MI455X (gfx1250) — matrix operand order repaired and checked by reading, not compiled, not run on hardware
//
#include <hip/hip_runtime.h>
#include <hip/hip_bf16.h>

typedef _Float16 h16;
typedef __attribute__((ext_vector_type(16))) _Float16 v16h;
typedef __attribute__((ext_vector_type(8)))  _Float16 v8h;
typedef __attribute__((ext_vector_type(8)))  float    v8f;
typedef __attribute__((ext_vector_type(4)))  unsigned int u32x4;
typedef __attribute__((ext_vector_type(4)))  int i32x4;
typedef __attribute__((ext_vector_type(8)))  int i32x8;

#define NB   128
#define RR   300
#define EMB  32

// --------------------------------------------------------------------------
// TDM feature detection: builtin exists on both toolchains but with different
// arity (5 on ROCm7.2/clang-22, 6 on amdgpu-toolchain/clang-23 whose headers
// ship amd_gfx1250_TDM.h). Fall back to a manual LDS copy if absent.
// --------------------------------------------------------------------------
#if defined(__has_builtin)
# if __has_builtin(__builtin_amdgcn_tensor_load_to_lds)
#  define HAVE_TDM 1
# endif
#endif
#if defined(__has_include)
# if __has_include(<hip/amd_detail/amd_gfx1250_TDM.h>)
#  define TDM_6ARG 1
# endif
#endif

#ifdef HAVE_TDM
__device__ __forceinline__ void tdm_issue(u32x4 g0, i32x8 g1) {
    i32x4 z4 = {0, 0, 0, 0};
#ifdef TDM_6ARG
    i32x8 z8 = {0, 0, 0, 0, 0, 0, 0, 0};
    __builtin_amdgcn_tensor_load_to_lds(g0, g1, z4, z4, z8, 0);
#else
    __builtin_amdgcn_tensor_load_to_lds(g0, g1, z4, z4, 0);
#endif
}
__device__ __forceinline__ void tdm_wait() {
#if __has_builtin(__builtin_amdgcn_s_wait_tensorcnt)
    __builtin_amdgcn_s_wait_tensorcnt((short)0);
#else
    asm volatile("s_wait_tensorcnt 0x0" ::: "memory");
#endif
    asm volatile("" ::: "memory");   // keep ds loads below the wait
}
#endif

// ---------------------------------------------------------------------------
// Weight fragment pre-swizzle: OIHW f32 -> per-(tap,kTile,coTile) B-fragments
// (ISA 16-bit B 32x16 layout: lane n = lane&15; lane half h owns K 8h..8h+7
// and 16+8h..16+8h+7). FC layers are the ksize==1 case (w = [Nout][Kin]).
// ---------------------------------------------------------------------------
__global__ void prep_wfrag(const float* __restrict__ w, h16* __restrict__ wf,
                           int ksize, int Ci, int KT, int COT, int total) {
    int i = blockIdx.x * blockDim.x + threadIdx.x;
    if (i >= total) return;
    int e    = i & 15;
    int lane = (i >> 4) & 31;
    int f    = i >> 9;                 // fragment id = (tap*KT + t)*COT + ct
    int ct   = f % COT;
    int t    = (f / COT) % KT;
    int tap  = f / (COT * KT);
    int ky = tap / ksize, kx = tap % ksize;
    int nn = lane & 15, hh = lane >> 4;
    int ci = t * 32 + (e >> 3) * 16 + hh * 8 + (e & 7);
    int co = ct * 16 + nn;
    float v = (ci < Ci) ? w[(((size_t)co * Ci + ci) * ksize + ky) * ksize + kx] : 0.f;
    wf[i] = (h16)v;
}

// ---------------------------------------------------------------------------
// Room scatter: atomic f32 adds into [N][72][72][33] grid (rooms may overlap).
// ---------------------------------------------------------------------------
__global__ void scatter_rooms(const int* __restrict__ X, const int* __restrict__ rm,
                              const float* __restrict__ emb, float* __restrict__ grid) {
    int nr = blockIdx.x;
    int n = nr / RR, r = nr % RR;
    int cell = threadIdx.x;            // 0..63
    int a = cell >> 3, b = cell & 7;   // a = width idx, b = height idx
    if (rm[(r * 8 + a) * 8 + b] == 0) return;
    int ix = X[((size_t)n * RR + r) * 2 + 0] + a;   // MAP_X coord
    int iy = X[((size_t)n * RR + r) * 2 + 1] + b;   // MAP_Y coord
    float* g = grid + (((size_t)n * 72 + iy) * 72 + ix) * 33;   // [n][y][x][c]
    atomicAdd(g, 1.0f);
    #pragma unroll 4
    for (int e = 0; e < EMB; ++e) atomicAdd(g + 1 + e, emb[(size_t)r * EMB + e]);
}

// ---------------------------------------------------------------------------
// f32 grid [N][72][72][33] -> f16 padded NHWC [N][76][76][64] (halo=2, C pad 64)
// ---------------------------------------------------------------------------
__global__ void grid_to_h16(const float* __restrict__ g32, h16* __restrict__ dst, int total) {
    int i = blockIdx.x * blockDim.x + threadIdx.x;
    if (i >= total) return;                      // total = N*76*76
    int px = i % 76, py = (i / 76) % 76, n = i / (76 * 76);
    h16* o = dst + (size_t)i * 64;
    int y = py - 2, x = px - 2;
    bool inr = (y >= 0 && y < 72 && x >= 0 && x < 72);
    const float* g = g32 + (((size_t)n * 72 + (inr ? y : 0)) * 72 + (inr ? x : 0)) * 33;
    #pragma unroll 8
    for (int c = 0; c < 64; ++c)
        o[c] = (inr && c < 33) ? (h16)g[c] : (h16)0.f;
}

// ---------------------------------------------------------------------------
// Implicit-GEMM conv via V_WMMA_F32_16X16X32_F16 with TDM-staged LDS patches.
// Input : f16 NHWC, halo-padded [N][HOUT+K-1][WOUT+K-1][CIN_P]
// Output: f32 NHWC [N][HOUT][WOUT][COUT], bias added (ReLU deferred to pool).
// One wave per (n, y, x-tile of 16 pixels); x-tiles overlap at the edge so all
// lanes are always valid (EXEC all-ones, required by WMMA). Each wave DMAs its
// KSIZE x (16+K-1) x CIN_P input patch into LDS once (TENSORcnt), then every
// filter tap reads A-fragments with ds_load_b128 (no redundant HBM traffic).
// ---------------------------------------------------------------------------
template<int KSIZE, int CIN_P, int COUT, int HOUT, int WOUT, int XTILES>
__global__ __launch_bounds__(128)
void conv_wmma(const h16* __restrict__ in, const h16* __restrict__ wf,
               const float* __restrict__ bias, float* __restrict__ out,
               int total_tiles) {
    constexpr int HIN_P  = HOUT + KSIZE - 1;
    constexpr int WIN_P  = WOUT + KSIZE - 1;
    constexpr int KT     = CIN_P / 32;
    constexpr int COT    = COUT / 16;
    constexpr int PATCHW = 16 + KSIZE - 1;
    constexpr int ROWH   = PATCHW * CIN_P;        // halfs per patch row
    constexpr int PATCH  = KSIZE * ROWH;          // halfs per wave patch
    __shared__ __attribute__((aligned(16))) h16 lds[4 * PATCH];

    const int wv  = threadIdx.x >> 5;
    const int wid = blockIdx.x * 4 + wv;
    if (wid >= total_tiles) return;
    const int lane = threadIdx.x & 31;
    const int tx   = wid % XTILES;
    const int y    = (wid / XTILES) % HOUT;
    const int n    = wid / (XTILES * HOUT);
    const int x0   = (tx * 16 + 16 <= WOUT) ? tx * 16 : (WOUT - 16);

    h16* lp = &lds[wv * PATCH];
    const h16* gp = in + (((size_t)n * HIN_P + y) * WIN_P + x0) * CIN_P;

#ifdef HAVE_TDM
    {   // D# per ISA ch.8: group0 = {ctrl, lds_addr, global_addr, type=2},
        // group1 = {mask|data_size, dims/tiles/strides} for a 2-D f16 tile.
        unsigned long long ga = (unsigned long long)(uintptr_t)gp;
        unsigned la = (unsigned)(uintptr_t)lp;
        u32x4 g0;
        g0.x = 1u;                                             // count=1, user D#
        g0.y = la;                                             // lds_addr (bytes)
        g0.z = (unsigned)ga;                                   // global_addr lo
        g0.w = ((unsigned)(ga >> 32) & 0x01FFFFFFu) | (2u << 30); // addr hi | type=2
        i32x8 g1;
        g1[0] = (int)(1u << 16);                               // data_size=2B, mask=0
        g1[1] = (int)0xFFFF0000u;                              // tensor_dim0 lo16 (big)
        g1[2] = (int)0xFFFF7FFFu;                              // dim0 hi16 | dim1 lo16
        g1[3] = (int)(((unsigned)ROWH << 16) | 0x7FFFu);       // dim1 hi16 | tile_dim0
        g1[4] = KSIZE;                                         // tile_dim1 (tile_dim2=0)
        g1[5] = WIN_P * CIN_P;                                 // tensor_dim0_stride lo
        g1[6] = 0;                                             // stride hi | dim1_stride lo
        g1[7] = 0;
        tdm_issue(g0, g1);
    }
#else
    {   // fallback: wave-cooperative copy, 16B per lane per step
        for (int r = 0; r < KSIZE; ++r)
            for (int j = lane * 8; j < ROWH; j += 32 * 8)
                *(v8h*)(lp + r * ROWH + j) = *(const v8h*)(gp + (size_t)r * WIN_P * CIN_P + j);
    }
#endif

    const int mlane = lane & 15;       // A: pixel row; D: co column
    const int khalf = lane >> 4;

#ifdef HAVE_TDM
    tdm_wait();
#endif

    for (int cc = 0; cc < COT; cc += 4) {
        v8f acc[4] = {};
        #pragma unroll
        for (int ky = 0; ky < KSIZE; ++ky) {
            #pragma unroll
            for (int kx = 0; kx < KSIZE; ++kx) {
                const int tapb = (ky * KSIZE + kx) * KT;
                const h16* ap = lp + ((size_t)ky * PATCHW + (mlane + kx)) * CIN_P + khalf * 8;
                #pragma unroll
                for (int t = 0; t < KT; ++t) {
                    v8h lo = *(const v8h*)(ap + t * 32);
                    v8h hi = *(const v8h*)(ap + t * 32 + 16);
                    v16h a = __builtin_shufflevector(lo, hi,
                        0,1,2,3,4,5,6,7,8,9,10,11,12,13,14,15);
                    const h16* bf = wf + ((size_t)(tapb + t) * COT + cc) * 512 + lane * 16;
                    #pragma unroll
                    for (int c = 0; c < 4; ++c) {
                        v16h b = *(const v16h*)(bf + c * 512);
                        acc[c] = __builtin_amdgcn_wmma_f32_16x16x32_f16(
                            false, a, false, b, (short)0, acc[c], false, false);
                    }
                }
            }
        }
        // D layout: column = lane&15, rows = (lane>>4)*8 + g
        const int mb = khalf * 8;
        #pragma unroll
        for (int c = 0; c < 4; ++c) {
            int co = (cc + c) * 16 + mlane;
            float bv = bias[co];
            #pragma unroll
            for (int g = 0; g < 8; ++g) {
                int xo = x0 + mb + g;
                out[(((size_t)n * HOUT + y) * WOUT + xo) * COUT + co] = acc[c][g] + bv;
            }
        }
    }
}

// ---------------------------------------------------------------------------
// 3x3 stride-2 VALID maxpool + ReLU, f32 NHWC in -> f16 NHWC out with halo=pad
// ---------------------------------------------------------------------------
__global__ void pool3s2_relu(const float* __restrict__ in, h16* __restrict__ out,
                             int Hi, int Wi, int C, int Ho, int Wo, int pad, int total) {
    int i = blockIdx.x * blockDim.x + threadIdx.x;
    if (i >= total) return;
    int Wp = Wo + 2 * pad, Hp = Ho + 2 * pad;
    int c  = i % C;
    int px = (i / C) % Wp;
    int py = (i / (C * Wp)) % Hp;
    int n  = i / (C * Wp * Hp);
    h16 r = (h16)0.f;
    int oy = py - pad, ox = px - pad;
    if (oy >= 0 && oy < Ho && ox >= 0 && ox < Wo) {
        float m = -__builtin_inff();
        #pragma unroll
        for (int dy = 0; dy < 3; ++dy)
            #pragma unroll
            for (int dx = 0; dx < 3; ++dx)
                m = fmaxf(m, in[(((size_t)n * Hi + (2 * oy + dy)) * Wi + (2 * ox + dx)) * C + c]);
        r = (h16)fmaxf(m, 0.f);
    }
    out[i] = r;
}

// ---------------------------------------------------------------------------
// Fused pool3 (17x17 -> 8x8) + ReLU + global average -> f16 [N][256] FC input.
// ---------------------------------------------------------------------------
__global__ void pool_avg(const float* __restrict__ in, h16* __restrict__ out, int total) {
    int i = blockIdx.x * blockDim.x + threadIdx.x;
    if (i >= total) return;                  // total = N*256
    int c = i % 256, n = i / 256;
    float s = 0.f;
    for (int oy = 0; oy < 8; ++oy)
        for (int ox = 0; ox < 8; ++ox) {
            float m = -__builtin_inff();
            #pragma unroll
            for (int dy = 0; dy < 3; ++dy)
                #pragma unroll
                for (int dx = 0; dx < 3; ++dx)
                    m = fmaxf(m, in[(((size_t)n * 17 + 2 * oy + dy) * 17 + 2 * ox + dx) * 256 + c]);
            s += fmaxf(m, 0.f);
        }
    out[i] = (h16)(s * (1.f / 64.f));
}

// ---------------------------------------------------------------------------
// FC layer on WMMA: out[m,n] = x[m,:] . w[n,:] + b[n]; M=128 (8 tiles).
// ---------------------------------------------------------------------------
template<bool RELU, bool OUT_HALF>
__global__ __launch_bounds__(256)
void fc_wmma(const h16* __restrict__ x, const h16* __restrict__ wf,
             const float* __restrict__ bias, float* __restrict__ out32,
             h16* __restrict__ out16, int Kin, int Nout, int total_tiles) {
    const int wid = blockIdx.x * (blockDim.x >> 5) + (threadIdx.x >> 5);
    if (wid >= total_tiles) return;
    const int lane = threadIdx.x & 31;
    const int NT = Nout / 16, KT = Kin / 32;
    const int nchunks = NT / 4;
    const int mt = wid / nchunks;
    const int nc = (wid % nchunks) * 4;
    const int mlane = lane & 15, khalf = lane >> 4;
    v8f acc[4] = {};
    const h16* xp = x + ((size_t)(mt * 16 + mlane)) * Kin + khalf * 8;
    for (int t = 0; t < KT; ++t) {
        v8h lo = *(const v8h*)(xp + t * 32);
        v8h hi = *(const v8h*)(xp + t * 32 + 16);
        v16h a = __builtin_shufflevector(lo, hi,
            0,1,2,3,4,5,6,7,8,9,10,11,12,13,14,15);
        const h16* bf = wf + ((size_t)t * NT + nc) * 512 + lane * 16;
        #pragma unroll
        for (int c = 0; c < 4; ++c) {
            v16h b = *(const v16h*)(bf + c * 512);
            acc[c] = __builtin_amdgcn_wmma_f32_16x16x32_f16(
                false, a, false, b, (short)0, acc[c], false, false);
        }
    }
    const int mb = khalf * 8;
    #pragma unroll
    for (int c = 0; c < 4; ++c) {
        int nn = (nc + c) * 16 + mlane;
        float bv = bias[nn];
        #pragma unroll
        for (int g = 0; g < 8; ++g) {
            int row = mt * 16 + mb + g;
            float v = acc[c][g] + bv;
            if (RELU) v = fmaxf(v, 0.f);
            if (OUT_HALF) out16[(size_t)row * Nout + nn] = (h16)v;
            else          out32[(size_t)row * Nout + nn] = v;
        }
    }
}

// ---------------------------------------------------------------------------
// Host orchestration
// ---------------------------------------------------------------------------
static inline size_t alup(size_t x) { return (x + 255) & ~(size_t)255; }
static inline unsigned nblk(long long t, int b) { return (unsigned)((t + b - 1) / b); }

extern "C" void kernel_launch(void* const* d_in, const int* in_sizes, int n_in,
                              void* d_out, int out_size, void* d_ws, size_t ws_size,
                              hipStream_t stream) {
    const int*   X   = (const int*)  d_in[0];
    const int*   rm  = (const int*)  d_in[1];
    const float* emb = (const float*)d_in[2];
    const float* cw0 = (const float*)d_in[3];
    const float* cb0 = (const float*)d_in[4];
    const float* cw1 = (const float*)d_in[5];
    const float* cb1 = (const float*)d_in[6];
    const float* cw2 = (const float*)d_in[7];
    const float* cb2 = (const float*)d_in[8];
    const float* fw0 = (const float*)d_in[9];
    const float* fb0 = (const float*)d_in[10];
    const float* fw1 = (const float*)d_in[11];
    const float* fb1 = (const float*)d_in[12];
    const float* fw2 = (const float*)d_in[13];
    const float* fb2 = (const float*)d_in[14];

    char* ws = (char*)d_ws;
    // Big aliased regions (A: grid / pool1-out / conv3-out, B: conv1-in /
    // conv2-out, C: conv1-out / pool2-out)
    const size_t SZ_A = alup((size_t)NB * 72 * 72 * 33 * 4);   // 87.6 MB
    const size_t SZ_B = alup((size_t)NB * 76 * 76 * 64 * 2);   // 94.6 MB
    const size_t SZ_C = alup((size_t)NB * 72 * 72 * 64 * 4);   // 169.9 MB
    size_t off = 0;
    char* regA = ws + off; off += SZ_A;
    char* regB = ws + off; off += SZ_B;
    char* regC = ws + off; off += SZ_C;
    float* g32   = (float*)regA;            // [N][72][72][33] f32
    h16*   c1in  = (h16*)  regB;            // [N][76][76][64] f16
    float* c1out = (float*)regC;            // [N][72][72][64] f32
    h16*   p1out = (h16*)  regA;            // [N][37][37][64] f16 (grid dead)
    float* c2out = (float*)regB;            // [N][35][35][128] f32 (c1in dead)
    h16*   p2out = (h16*)  regC;            // [N][19][19][128] f16 (c1out dead)
    float* c3out = (float*)regA;            // [N][17][17][256] f32 (p1out dead)

    h16* wf1  = (h16*)(ws + off); off += alup((size_t)25 * 2 * 4  * 512 * 2);
    h16* wf2  = (h16*)(ws + off); off += alup((size_t)9  * 2 * 8  * 512 * 2);
    h16* wf3  = (h16*)(ws + off); off += alup((size_t)9  * 4 * 16 * 512 * 2);
    h16* wff0 = (h16*)(ws + off); off += alup((size_t)8  * 32 * 512 * 2);
    h16* wff1 = (h16*)(ws + off); off += alup((size_t)16 * 16 * 512 * 2);
    h16* wff2 = (h16*)(ws + off); off += alup((size_t)8  * 32 * 512 * 2);
    h16* avg16 = (h16*)(ws + off); off += alup((size_t)NB * 256 * 2);
    h16* fc0o  = (h16*)(ws + off); off += alup((size_t)NB * 512 * 2);
    h16* fc1o  = (h16*)(ws + off); off += alup((size_t)NB * 256 * 2);
    (void)ws_size; (void)in_sizes; (void)n_in; (void)out_size;

    // Weight fragment swizzles (cheap; rerun every call for determinism)
    {
        int t;
        t = 25 * 2 * 4  * 512; prep_wfrag<<<nblk(t,256),256,0,stream>>>(cw0, wf1, 5, 33,  2, 4,  t);
        t = 9  * 2 * 8  * 512; prep_wfrag<<<nblk(t,256),256,0,stream>>>(cw1, wf2, 3, 64,  2, 8,  t);
        t = 9  * 4 * 16 * 512; prep_wfrag<<<nblk(t,256),256,0,stream>>>(cw2, wf3, 3, 128, 4, 16, t);
        t = 8  * 32 * 512;     prep_wfrag<<<nblk(t,256),256,0,stream>>>(fw0, wff0, 1, 256, 8,  32, t);
        t = 16 * 16 * 512;     prep_wfrag<<<nblk(t,256),256,0,stream>>>(fw1, wff1, 1, 512, 16, 16, t);
        t = 8  * 32 * 512;     prep_wfrag<<<nblk(t,256),256,0,stream>>>(fw2, wff2, 1, 256, 8,  32, t);
    }

    // Stage 0: scatter rooms into f32 grid, then convert to padded f16 NHWC
    hipMemsetAsync(g32, 0, (size_t)NB * 72 * 72 * 33 * 4, stream);
    scatter_rooms<<<NB * RR, 64, 0, stream>>>(X, rm, emb, g32);
    {
        int t = NB * 76 * 76;
        grid_to_h16<<<nblk(t,256),256,0,stream>>>(g32, c1in, t);
    }

    // Stage 1: conv1 (33->64, k5) + pool
    conv_wmma<5, 64, 64, 72, 72, 5><<<nblk((long long)NB*72*5, 4), 128, 0, stream>>>(
        c1in, wf1, cb0, c1out, NB * 72 * 5);
    {
        int t = NB * 37 * 37 * 64;
        pool3s2_relu<<<nblk(t,256),256,0,stream>>>(c1out, p1out, 72, 72, 64, 35, 35, 1, t);
    }

    // Stage 2: conv2 (64->128, k3) + pool
    conv_wmma<3, 64, 128, 35, 35, 3><<<nblk((long long)NB*35*3, 4), 128, 0, stream>>>(
        p1out, wf2, cb1, c2out, NB * 35 * 3);
    {
        int t = NB * 19 * 19 * 128;
        pool3s2_relu<<<nblk(t,256),256,0,stream>>>(c2out, p2out, 35, 35, 128, 17, 17, 1, t);
    }

    // Stage 3: conv3 (128->256, k3) + fused pool/avg
    conv_wmma<3, 128, 256, 17, 17, 2><<<nblk((long long)NB*17*2, 4), 128, 0, stream>>>(
        p2out, wf3, cb2, c3out, NB * 17 * 2);
    {
        int t = NB * 256;
        pool_avg<<<nblk(t,256),256,0,stream>>>(c3out, avg16, t);
    }

    // FC stack on WMMA
    fc_wmma<true,  true ><<<nblk(8 * (512/16/4), 8), 256, 0, stream>>>(
        avg16, wff0, fb0, nullptr, fc0o, 256, 512, 8 * (512/16/4));
    fc_wmma<true,  true ><<<nblk(8 * (256/16/4), 8), 256, 0, stream>>>(
        fc0o, wff1, fb1, nullptr, fc1o, 512, 256, 8 * (256/16/4));
    fc_wmma<false, false><<<nblk(8 * (512/16/4), 8), 256, 0, stream>>>(
        fc1o, wff2, fb2, (float*)d_out, nullptr, 256, 512, 8 * (512/16/4));
}
